// SelfAttention_83167746720451
// MI455X (gfx1250) — matrix operand order repaired and checked by reading, not compiled, not run on hardware
//
#include <hip/hip_runtime.h>

// ---------------- types ----------------
typedef __bf16 bf16;
typedef __attribute__((ext_vector_type(16))) __bf16 v16bf;
typedef __attribute__((ext_vector_type(8)))  __bf16 bf16x8;
typedef __attribute__((ext_vector_type(4)))  __bf16 bf16x4;
typedef __attribute__((ext_vector_type(8)))  float  v8f;

#define T_DIM 2048
#define C_DIM 1024
#define H_NUM 16
#define H_DIM 64
#define SCALE_Q 0.125f   // 64^-0.5

__device__ __forceinline__ v8f zero_v8f() {
  v8f z;
#pragma unroll
  for (int i = 0; i < 8; ++i) z[i] = 0.0f;
  return z;
}

// A-operand fragment: 16x32 bf16, row-major source, K interleaved per ISA
// lane<16: row=lane, K chunks {0..7, 16..23}; lane>=16: row=lane-16, K {8..15, 24..31}
__device__ __forceinline__ v16bf frag_a(const bf16* tile, int ld) {
  int lane = threadIdx.x & 31;
  const bf16* p = tile + (lane & 15) * ld + ((lane >> 4) << 3);
  bf16x8 lo = *(const bf16x8*)p;
  bf16x8 hi = *(const bf16x8*)(p + 16);
  return __builtin_shufflevector(lo, hi, 0,1,2,3,4,5,6,7,8,9,10,11,12,13,14,15);
}

// B-operand fragment: 32x16 bf16; source stored "B^T row-major" (row n = 32
// contiguous K values). Same K interleave as A: lane<16 col=lane K {0..7,16..23},
// lane>=16 col=lane-16 K {8..15,24..31} -> two 16B reads per lane.
__device__ __forceinline__ v16bf frag_b(const bf16* tileT, int ld) {
  int lane = threadIdx.x & 31;
  const bf16* p = tileT + (lane & 15) * ld + ((lane >> 4) << 3);
  bf16x8 lo = *(const bf16x8*)p;
  bf16x8 hi = *(const bf16x8*)(p + 16);
  return __builtin_shufflevector(lo, hi, 0,1,2,3,4,5,6,7,8,9,10,11,12,13,14,15);
}

__device__ __forceinline__ v8f wmma_bf16(v16bf a, v16bf b, v8f c) {
  return __builtin_amdgcn_wmma_f32_16x16x32_bf16(false, a, false, b, (short)0, c,
                                                 false, false);
}

// --------- CDNA5 async copy: global -> LDS (tracked by ASYNCcnt) ---------
// per-wave instruction: each lane moves 16B.  %0 = LDS byte address (VGPR),
// %1 = 64-bit global address (VGPR pair).
__device__ __forceinline__ void async_cp_16B(unsigned lds_addr, const void* gptr) {
  asm volatile("global_load_async_to_lds_b128 %0, %1, off"
               :: "v"(lds_addr), "v"(gptr) : "memory");
}
__device__ __forceinline__ void wait_async0() {
  asm volatile("s_wait_asynccnt 0x0" ::: "memory");
}

// ---------------- kernel 0: f32 -> bf16 conversion ----------------
__global__ void __launch_bounds__(256) cvt_f32_bf16(const float* __restrict__ src,
                                                    bf16* __restrict__ dst, int n4) {
  int i = blockIdx.x * blockDim.x + threadIdx.x;
  if (i < n4) {
    float4 f = reinterpret_cast<const float4*>(src)[i];
    bf16x4 o;
    o[0] = (bf16)f.x; o[1] = (bf16)f.y; o[2] = (bf16)f.z; o[3] = (bf16)f.w;
    reinterpret_cast<bf16x4*>(dst)[i] = o;
  }
}

// ---------------- shared 64x64-per-wave GEMM mainloop ----------------
template <int KDIM>
__device__ __forceinline__ void gemm_wave_64x64(const bf16* __restrict__ A,
                                                const bf16* __restrict__ Bt,
                                                int lda, int ldb,
                                                int row0, int col0, v8f acc[4][4]) {
#pragma unroll
  for (int i = 0; i < 4; ++i)
#pragma unroll
    for (int j = 0; j < 4; ++j) acc[i][j] = zero_v8f();

  for (int k0 = 0; k0 < KDIM; k0 += 32) {
    v16bf af[4], bfr[4];
#pragma unroll
    for (int mt = 0; mt < 4; ++mt)
      af[mt] = frag_a(A + (row0 + 16 * mt) * lda + k0, lda);
#pragma unroll
    for (int nt = 0; nt < 4; ++nt)
      bfr[nt] = frag_b(Bt + (col0 + 16 * nt) * ldb + k0, ldb);
#pragma unroll
    for (int mt = 0; mt < 4; ++mt)
#pragma unroll
      for (int nt = 0; nt < 4; ++nt)
        acc[mt][nt] = wmma_bf16(af[mt], bfr[nt], acc[mt][nt]);
  }
}

// ---------------- kernel 1: QKV projection + scatter ----------------
__global__ void __launch_bounds__(256) qkv_gemm(const bf16* __restrict__ xb,
                                                const bf16* __restrict__ wb,
                                                const float* __restrict__ bias,
                                                bf16* __restrict__ q,
                                                bf16* __restrict__ k,
                                                bf16* __restrict__ vT) {
  int warp = threadIdx.x >> 5;
  int wid = blockIdx.x * 8 + warp;
  const int TN = (3 * C_DIM) / 64;  // 48
  int row0 = (wid / TN) * 64, col0 = (wid % TN) * 64;

  v8f acc[4][4];
  gemm_wave_64x64<C_DIM>(xb, wb, C_DIM, C_DIM, row0, col0, acc);

  int lane = threadIdx.x & 31, half = lane >> 4, nl = lane & 15;
#pragma unroll
  for (int mt = 0; mt < 4; ++mt)
#pragma unroll
    for (int nt = 0; nt < 4; ++nt) {
      int n = col0 + nt * 16 + nl;
      float bv = bias[n];
      int sel = n >> 10, cc = n & 1023, h = cc >> 6, d = cc & 63;
#pragma unroll
      for (int r = 0; r < 8; ++r) {
        int m = row0 + mt * 16 + r + 8 * half;
        int b = m >> 11, t = m & 2047;
        int bh = b * H_NUM + h;
        float v = acc[mt][nt][r] + bv;
        if (sel == 0)      q[(bh * T_DIM + t) * H_DIM + d] = (bf16)(v * SCALE_Q);
        else if (sel == 1) k[(bh * T_DIM + t) * H_DIM + d] = (bf16)v;
        else               vT[(bh * H_DIM + d) * T_DIM + t] = (bf16)v;
      }
    }
}

// ---------------- kernel 2: causal flash attention ----------------
// block = 8 waves = 8 adjacent 32-query tiles of ONE head; block walks key
// tiles in lockstep; K/V tiles double-buffered in LDS via async copies.
__global__ void __launch_bounds__(256) flash_attn(const bf16* __restrict__ q,
                                                  const bf16* __restrict__ k,
                                                  const bf16* __restrict__ vT,
                                                  bf16* __restrict__ o) {
  __shared__ __align__(64) bf16 Plds[8][1024];   // per-wave 32x32 P tile
  __shared__ __align__(64) bf16 Kst[2][2048];    // 32 keys x 64 feat
  __shared__ __align__(64) bf16 Vst[2][2048];    // 64 feat x 32 keys

  int tid = threadIdx.x;
  int warp = tid >> 5;
  int lane = tid & 31, half = lane >> 4, nl = lane & 15;
  int bh = blockIdx.x >> 3;        // head index (0..31)
  int slab = blockIdx.x & 7;       // 8 query-tile slabs per head
  int qt = slab * 8 + warp;        // this wave's 32-query tile
  int q0 = qt * 32;

  const bf16* qh = q + (size_t)bh * T_DIM * H_DIM;
  const bf16* kh = k + (size_t)bh * T_DIM * H_DIM;
  const bf16* vh = vT + (size_t)bh * H_DIM * T_DIM;

  // resident Q fragments (pre-scaled by 1/sqrt(hd) in qkv_gemm)
  v16bf qf[2][2];
#pragma unroll
  for (int mt = 0; mt < 2; ++mt)
#pragma unroll
    for (int ks = 0; ks < 2; ++ks)
      qf[mt][ks] = frag_a(qh + (q0 + mt * 16) * H_DIM + ks * 32, H_DIM);

  v8f oacc[2][4];
#pragma unroll
  for (int mt = 0; mt < 2; ++mt)
#pragma unroll
    for (int nf = 0; nf < 4; ++nf) oacc[mt][nf] = zero_v8f();
  float m_run[2][8], l_run[2][8];
#pragma unroll
  for (int mt = 0; mt < 2; ++mt)
#pragma unroll
    for (int r = 0; r < 8; ++r) { m_run[mt][r] = -1e30f; l_run[mt][r] = 0.0f; }

  bf16* pw = Plds[warp];
  int my_kt_max = qt;              // causal: last key tile = own query tile
  int blk_kt_max = slab * 8 + 7;   // block-wide lockstep bound

  // cooperative async stage of K/V tile kt into buffer b:
  // K tile: 4KB contiguous; V tile: 64 rows x 64B strided from v^T
  auto stage = [&](int kt, int b) {
    async_cp_16B((unsigned)(unsigned long long)(&Kst[b][tid * 8]),
                 (const char*)(kh + kt * 2048) + tid * 16);
    int row = tid >> 2, cp = tid & 3;
    async_cp_16B((unsigned)(unsigned long long)(&Vst[b][row * 32 + cp * 8]),
                 (const char*)(vh + row * T_DIM + kt * 32) + cp * 16);
  };

  stage(0, 0);
  wait_async0();
  __syncthreads();

  for (int kt = 0; kt <= blk_kt_max; ++kt) {
    int cur = kt & 1;
    if (kt < blk_kt_max) stage(kt + 1, cur ^ 1);   // async prefetch next tile

    if (kt <= my_kt_max) {                         // wave-uniform predicate
      const bf16* Kc = Kst[cur];
      const bf16* Vc = Vst[cur];

      // S = Qs * K^T  (32 queries x 32 keys), K from LDS
      v8f s[2][2];
#pragma unroll
      for (int mt = 0; mt < 2; ++mt)
#pragma unroll
        for (int nt = 0; nt < 2; ++nt) s[mt][nt] = zero_v8f();
      v16bf kf[2][2];
#pragma unroll
      for (int nt = 0; nt < 2; ++nt)
#pragma unroll
        for (int ks = 0; ks < 2; ++ks)
          kf[nt][ks] = frag_b(Kc + (nt * 16) * H_DIM + ks * 32, H_DIM);
#pragma unroll
      for (int mt = 0; mt < 2; ++mt)
#pragma unroll
        for (int nt = 0; nt < 2; ++nt)
#pragma unroll
          for (int ks = 0; ks < 2; ++ks)
            s[mt][nt] = wmma_bf16(qf[mt][ks], kf[nt][ks], s[mt][nt]);

      if (kt == my_kt_max) {  // diagonal tile: causal mask
#pragma unroll
        for (int mt = 0; mt < 2; ++mt)
#pragma unroll
          for (int nt = 0; nt < 2; ++nt)
#pragma unroll
            for (int r = 0; r < 8; ++r) {
              int kg = kt * 32 + nt * 16 + nl;
              int qg = q0 + mt * 16 + r + 8 * half;
              if (kg > qg) s[mt][nt][r] = -1e30f;
            }
      }

      // online softmax per 16-row tile
#pragma unroll
      for (int mt = 0; mt < 2; ++mt) {
        float mloc[8];
#pragma unroll
        for (int r = 0; r < 8; ++r) mloc[r] = fmaxf(s[mt][0][r], s[mt][1][r]);
#pragma unroll
        for (int off = 1; off < 16; off <<= 1)
#pragma unroll
          for (int r = 0; r < 8; ++r)
            mloc[r] = fmaxf(mloc[r], __shfl_xor(mloc[r], off, 32));
#pragma unroll
        for (int r = 0; r < 8; ++r) {
          float mn = fmaxf(m_run[mt][r], mloc[r]);
          float sc = __expf(m_run[mt][r] - mn);
          m_run[mt][r] = mn;
          float p0 = __expf(s[mt][0][r] - mn);
          float p1 = __expf(s[mt][1][r] - mn);
          s[mt][0][r] = p0; s[mt][1][r] = p1;
          float rs = p0 + p1;
#pragma unroll
          for (int off = 1; off < 16; off <<= 1) rs += __shfl_xor(rs, off, 32);
          l_run[mt][r] = l_run[mt][r] * sc + rs;
#pragma unroll
          for (int nf = 0; nf < 4; ++nf) oacc[mt][nf][r] *= sc;
        }
        // P -> LDS (row-major 32x32) for layout conversion C->A
#pragma unroll
        for (int nt = 0; nt < 2; ++nt)
#pragma unroll
          for (int r = 0; r < 8; ++r) {
            int row = mt * 16 + r + 8 * half;
            int col = nt * 16 + nl;
            pw[row * 32 + col] = (bf16)s[mt][nt][r];
          }
      }

      // O += P * V   (A from LDS, B = V^T tile from LDS)
      v16bf pf[2];
#pragma unroll
      for (int mt = 0; mt < 2; ++mt) pf[mt] = frag_a(pw + mt * 16 * 32, 32);
#pragma unroll
      for (int nf = 0; nf < 4; ++nf) {
        v16bf vf = frag_b(Vc + (nf * 16) * 32, 32);
#pragma unroll
        for (int mt = 0; mt < 2; ++mt)
          oacc[mt][nf] = wmma_bf16(pf[mt], vf, oacc[mt][nf]);
      }
    }

    wait_async0();     // prefetch of kt+1 complete (issuer-side)
    __syncthreads();   // everyone done reading buffer `cur`
  }

  // normalize and store to (B,T,C) bf16
  int b = bh >> 4, h = bh & 15;
#pragma unroll
  for (int mt = 0; mt < 2; ++mt)
#pragma unroll
    for (int r = 0; r < 8; ++r) {
      float inv = 1.0f / l_run[mt][r];
      int t = q0 + mt * 16 + r + 8 * half;
#pragma unroll
      for (int nf = 0; nf < 4; ++nf) {
        int cidx = h * H_DIM + nf * 16 + nl;
        o[((size_t)b * T_DIM + t) * C_DIM + cidx] = (bf16)(oacc[mt][nf][r] * inv);
      }
    }
}

// ---------------- kernel 3: output projection ----------------
__global__ void __launch_bounds__(256) out_gemm(const bf16* __restrict__ ob,
                                                const bf16* __restrict__ wb,
                                                const float* __restrict__ bias,
                                                float* __restrict__ out) {
  int warp = threadIdx.x >> 5;
  int wid = blockIdx.x * 8 + warp;
  const int TN = C_DIM / 64;  // 16
  int row0 = (wid / TN) * 64, col0 = (wid % TN) * 64;

  v8f acc[4][4];
  gemm_wave_64x64<C_DIM>(ob, wb, C_DIM, C_DIM, row0, col0, acc);

  int lane = threadIdx.x & 31, half = lane >> 4, nl = lane & 15;
#pragma unroll
  for (int mt = 0; mt < 4; ++mt)
#pragma unroll
    for (int nt = 0; nt < 4; ++nt) {
      int n = col0 + nt * 16 + nl;
      float bv = bias[n];
#pragma unroll
      for (int r = 0; r < 8; ++r) {
        int m = row0 + mt * 16 + r + 8 * half;
        out[(size_t)m * C_DIM + n] = acc[mt][nt][r] + bv;
      }
    }
}

// ---------------- launcher ----------------
extern "C" void kernel_launch(void* const* d_in, const int* in_sizes, int n_in,
                              void* d_out, int out_size, void* d_ws, size_t ws_size,
                              hipStream_t stream) {
  const float* x    = (const float*)d_in[0];
  const float* Wqkv = (const float*)d_in[1];
  const float* bqkv = (const float*)d_in[2];
  const float* Wout = (const float*)d_in[3];
  const float* bout = (const float*)d_in[4];
  float* out = (float*)d_out;

  char* ws = (char*)d_ws;
  bf16* xb    = (bf16*)(ws + 0);           // 4096*1024*2  = 8 MiB
  bf16* wqkvb = (bf16*)(ws + 8388608);     // 3072*1024*2  = 6 MiB
  bf16* woutb = (bf16*)(ws + 14680064);    // 1024*1024*2  = 2 MiB
  bf16* qb    = (bf16*)(ws + 16777216);    // (B,H,T,hd)   = 8 MiB
  bf16* kb    = (bf16*)(ws + 25165824);    // (B,H,T,hd)   = 8 MiB
  bf16* vTb   = (bf16*)(ws + 33554432);    // (B,H,hd,T)   = 8 MiB
  bf16* ob    = (bf16*)(ws + 41943040);    // (B,T,C)      = 8 MiB

  cvt_f32_bf16<<<(1048576 + 255) / 256, 256, 0, stream>>>(x, xb, 1048576);
  cvt_f32_bf16<<<(786432 + 255) / 256, 256, 0, stream>>>(Wqkv, wqkvb, 786432);
  cvt_f32_bf16<<<(262144 + 255) / 256, 256, 0, stream>>>(Wout, woutb, 262144);

  qkv_gemm<<<384, 256, 0, stream>>>(xb, wqkvb, bqkv, qb, kb, vTb);
  flash_attn<<<256, 256, 0, stream>>>(qb, kb, vTb, ob);
  out_gemm<<<128, 256, 0, stream>>>(ob, woutb, bout, out);
}
